// FixedTopKLoRALinear_17944373363237
// MI455X (gfx1250) — compile-verified, matrix operand order checked by reading, not run on hardware
//
#include <hip/hip_runtime.h>
#include <hip/hip_bf16.h>

// ---------------------------------------------------------------------------
// Problem constants (from reference)
// ---------------------------------------------------------------------------
#define MDIM   8192          // B_*S = 4*2048
#define DDIM   4096          // D_IN
#define NDIM   4096          // D_OUT
#define RDIM   64            // LoRA rank
#define KTOP   16
#define INV_TEMP 10.0f       // 1/0.1
#define SCALE_V  1.0f        // ALPHA/K = 16/16

// Tile config for main GEMM: block 256x128, 8 waves as 4(M) x 2(N), wave 64x64
#define BM 256
#define BN 128
#define BK 32
#define LDP 40               // LDS row pitch in bf16 elems (80B, 16B-aligned rows)

typedef __bf16 v4bf  __attribute__((ext_vector_type(4)));
typedef __bf16 v8bf  __attribute__((ext_vector_type(8)));
typedef __bf16 v16bf __attribute__((ext_vector_type(16)));
typedef float  v8f   __attribute__((ext_vector_type(8)));

// f32x4 -> bf16x4 using native conversions (backend selects v_cvt*_bf16_f32)
static __device__ __forceinline__ v4bf cvt4(float4 v) {
    v4bf r;
    r.x = (__bf16)v.x; r.y = (__bf16)v.y; r.z = (__bf16)v.z; r.w = (__bf16)v.w;
    return r;
}
static __device__ __forceinline__ unsigned short f2bf1(float f) {
    union { __bf16 h; unsigned short u; } c;
    c.h = (__bf16)f;
    return c.u;
}

// Load one 16-bit A/B-operand fragment (16x32 bf16) for this lane from LDS.
// Per ISA layout: lane L: M/N = L&15, holds K = {kh..kh+7} and {kh+16..kh+23}
// with kh = (L>=16 ? 8 : 0).  p points at row start + kh; halves are 32B apart.
static __device__ __forceinline__ v16bf ldfrag(const unsigned short* p) {
    v8bf lo = *(const v8bf*)(p);
    v8bf hi = *(const v8bf*)(p + 16);
    return __builtin_shufflevector(lo, hi, 0,1,2,3,4,5,6,7,8,9,10,11,12,13,14,15);
}

// One BK=32 K-step for a wave tile of 64x64 (4x4 of 16x16 wmma tiles):
// 16 WMMAs per 8 fragment loads (16 ds_load_b128).
static __device__ __forceinline__ void compute_tile(const unsigned short* xsB,
                                                    const unsigned short* wsB,
                                                    int rA, int rB, int lane,
                                                    v8f (&acc)[4][4]) {
    const int l15 = lane & 15;
    const int kh  = (lane >> 4) << 3;
    v16bf af[4];
#pragma unroll
    for (int mi = 0; mi < 4; ++mi)
        af[mi] = ldfrag(xsB + (rA + mi * 16 + l15) * LDP + kh);
#pragma unroll
    for (int ni = 0; ni < 4; ++ni) {
        v16bf bfv = ldfrag(wsB + (rB + ni * 16 + l15) * LDP + kh);
#pragma unroll
        for (int mi = 0; mi < 4; ++mi)
            acc[mi][ni] = __builtin_amdgcn_wmma_f32_16x16x32_bf16(
                false, af[mi], false, bfv, (short)0, acc[mi][ni], false, false);
    }
}

// ---------------------------------------------------------------------------
// Kernel 1: z = x @ A^T, soft-top-k mask, write z_sparse*SCALE as bf16.
// Grid: MDIM/128 blocks, 256 threads (8 waves). Each wave: 16 rows x 64 cols.
// ---------------------------------------------------------------------------
__global__ void __launch_bounds__(256)
zk(const float* __restrict__ x, const float* __restrict__ Am,
   unsigned short* __restrict__ zsp) {
    __shared__ __align__(16) unsigned short xs[128 * LDP];
    __shared__ __align__(16) unsigned short as_[64 * LDP];
    __shared__ __align__(16) float zb[128 * 68];
    __shared__ __align__(16) float ab[128 * 68];

    const int tid  = threadIdx.x;
    const int lane = tid & 31;
    const int w    = tid >> 5;
    const int m0   = blockIdx.x * 128;
    const int rowq = tid >> 3;            // 32 rows per pass, 8 float4 per row
    const int colq = (tid & 7) * 4;

    v8f acc[4] = {};

    for (int k = 0; k < DDIM; k += BK) {
        __syncthreads();
        // stage x tile 128x32 (f32 -> bf16)
#pragma unroll
        for (int i = 0; i < 4; ++i) {
            int r = rowq + 32 * i;
            float4 v = *(const float4*)(x + (size_t)(m0 + r) * DDIM + k + colq);
            *(v4bf*)(&xs[r * LDP + colq]) = cvt4(v);
        }
        // stage A tile 64x32
#pragma unroll
        for (int i = 0; i < 2; ++i) {
            int r = rowq + 32 * i;
            float4 v = *(const float4*)(Am + (size_t)r * DDIM + k + colq);
            *(v4bf*)(&as_[r * LDP + colq]) = cvt4(v);
        }
        __syncthreads();

        const int l15 = lane & 15;
        const int kh  = (lane >> 4) << 3;
        v16bf af = ldfrag(&xs[(w * 16 + l15) * LDP + kh]);
#pragma unroll
        for (int ni = 0; ni < 4; ++ni) {
            v16bf bfv = ldfrag(&as_[(ni * 16 + l15) * LDP + kh]);
            acc[ni] = __builtin_amdgcn_wmma_f32_16x16x32_bf16(
                false, af, false, bfv, (short)0, acc[ni], false, false);
        }
    }

    // dump z tile to LDS (C-layout: lane = col, vgpr = row)
    {
        const int l15 = lane & 15;
        const int m8  = (lane >> 4) * 8;
#pragma unroll
        for (int ni = 0; ni < 4; ++ni)
#pragma unroll
            for (int r = 0; r < 8; ++r) {
                float zv = acc[ni][r];
                zb[(w * 16 + m8 + r) * 68 + ni * 16 + l15] = zv;
                ab[(w * 16 + m8 + r) * 68 + ni * 16 + l15] = fabsf(zv);
            }
    }
    __syncthreads();

    // soft top-k per row: threshold = 16th largest of |z| (rank-count select)
    if (tid < 128) {
        const int row = tid;
        const float* arow = &ab[row * 68];
        float thr = 3.4e38f;
        for (int j = 0; j < 64; ++j) {
            float aj = arow[j];
            int cnt = 0;
            for (int i = 0; i < 64; ++i) cnt += (arow[i] > aj) ? 1 : 0;
            if (cnt <= (KTOP - 1) && aj < thr) thr = aj;
        }
        for (int j = 0; j < 64; ++j) {
            float zv = zb[row * 68 + j];
            float m  = 1.0f / (1.0f + __expf(-(arow[j] - thr) * INV_TEMP));
            zsp[(size_t)(m0 + row) * RDIM + j] = f2bf1(zv * m * SCALE_V);
        }
    }
}

// ---------------------------------------------------------------------------
// Kernel 2: convert LoRA-up matrix B (NDIM x RDIM f32) to bf16.
// ---------------------------------------------------------------------------
__global__ void __launch_bounds__(256)
bconv(const float* __restrict__ Bm, unsigned short* __restrict__ Bbf) {
    int i = (blockIdx.x * 256 + threadIdx.x) * 4;
    if (i < NDIM * RDIM) {
        float4 v = *(const float4*)(Bm + i);
        *(v4bf*)(Bbf + i) = cvt4(v);
    }
}

// ---------------------------------------------------------------------------
// Kernel 3: fused out = x@W^T + bias + zsp@Bbf^T
// Grid: (NDIM/BN, MDIM/BM), 256 threads = 8 waves (4 M x 2 N), wave tile 64x64.
// Double-buffered LDS, register-prefetched global loads, on-the-fly f32->bf16.
// ---------------------------------------------------------------------------
__global__ void __launch_bounds__(256)
gemmk(const float* __restrict__ x, const float* __restrict__ W,
      const float* __restrict__ bias,
      const unsigned short* __restrict__ zsp,
      const unsigned short* __restrict__ Bbf,
      float* __restrict__ out) {
    __shared__ __align__(16) unsigned short xs[2][BM * LDP];   // 2 x 20 KB
    __shared__ __align__(16) unsigned short wl[2][BN * LDP];   // 2 x 10 KB

    const int tid   = threadIdx.x;
    const int lane  = tid & 31;
    const int w     = tid >> 5;
    const int waveM = w >> 1;           // 0..3 -> row offset *64
    const int waveN = w & 1;            // 0..1 -> col offset *64
    const int rA    = waveM * 64;
    const int rB    = waveN * 64;
    const int m0    = blockIdx.y * BM;
    const int n0    = blockIdx.x * BN;

    const int rowq = tid >> 3;          // staging: 32 rows/pass, 8 f4 per row
    const int colq = (tid & 7) * 4;

    v8f acc[4][4] = {};
    float4 xr[8], wr[4];

    // ---- initial global load (k=0) + stage into buffer 0 ----
#pragma unroll
    for (int i = 0; i < 8; ++i)
        xr[i] = *(const float4*)(x + (size_t)(m0 + rowq + 32 * i) * DDIM + colq);
#pragma unroll
    for (int i = 0; i < 4; ++i)
        wr[i] = *(const float4*)(W + (size_t)(n0 + rowq + 32 * i) * DDIM + colq);
#pragma unroll
    for (int i = 0; i < 8; ++i)
        *(v4bf*)(&xs[0][(rowq + 32 * i) * LDP + colq]) = cvt4(xr[i]);
#pragma unroll
    for (int i = 0; i < 4; ++i)
        *(v4bf*)(&wl[0][(rowq + 32 * i) * LDP + colq]) = cvt4(wr[i]);
    __syncthreads();

    int buf = 0;
    // ---- main K loop over D_IN ----
    for (int kt = 1; kt < DDIM / BK; ++kt) {
        const int k = kt * BK;
#pragma unroll
        for (int i = 0; i < 8; ++i)
            xr[i] = *(const float4*)(x + (size_t)(m0 + rowq + 32 * i) * DDIM + k + colq);
#pragma unroll
        for (int i = 0; i < 4; ++i)
            wr[i] = *(const float4*)(W + (size_t)(n0 + rowq + 32 * i) * DDIM + k + colq);

        compute_tile(xs[buf], wl[buf], rA, rB, lane, acc);

#pragma unroll
        for (int i = 0; i < 8; ++i)
            *(v4bf*)(&xs[buf ^ 1][(rowq + 32 * i) * LDP + colq]) = cvt4(xr[i]);
#pragma unroll
        for (int i = 0; i < 4; ++i)
            *(v4bf*)(&wl[buf ^ 1][(rowq + 32 * i) * LDP + colq]) = cvt4(wr[i]);
        __syncthreads();
        buf ^= 1;
    }
    compute_tile(xs[buf], wl[buf], rA, rB, lane, acc);
    __syncthreads();

    // ---- LoRA epilogue: accumulate zsp(M x 64) @ Bbf(N x 64)^T, K=64 ----
    {
        const int row2  = tid >> 2;          // 64 rows/pass, 4x uint4 per row
        const int colq2 = (tid & 3) * 8;     // in bf16 elems (16B chunks)
#pragma unroll
        for (int kk = 0; kk < 2; ++kk) {
#pragma unroll
            for (int i = 0; i < 4; ++i) {    // 256 rows of zsp tile
                int r = row2 + 64 * i;
                *(uint4*)(&xs[0][r * LDP + colq2]) =
                    *(const uint4*)(zsp + (size_t)(m0 + r) * RDIM + kk * 32 + colq2);
            }
#pragma unroll
            for (int i = 0; i < 2; ++i) {    // 128 rows of Bbf tile
                int r = row2 + 64 * i;
                *(uint4*)(&wl[0][r * LDP + colq2]) =
                    *(const uint4*)(Bbf + (size_t)(n0 + r) * RDIM + kk * 32 + colq2);
            }
            __syncthreads();
            compute_tile(xs[0], wl[0], rA, rB, lane, acc);
            __syncthreads();
        }
    }

    // ---- epilogue: bias + store ----
    const int l15 = lane & 15;
    const int m8  = (lane >> 4) * 8;
#pragma unroll
    for (int ni = 0; ni < 4; ++ni) {
        int gn = n0 + rB + ni * 16 + l15;
        float bv = bias[gn];
#pragma unroll
        for (int mi = 0; mi < 4; ++mi) {
            int gmB = m0 + rA + mi * 16 + m8;
#pragma unroll
            for (int r = 0; r < 8; ++r)
                out[(size_t)(gmB + r) * NDIM + gn] = acc[mi][ni][r] + bv;
        }
    }
}

// ---------------------------------------------------------------------------
// Host launch
// ---------------------------------------------------------------------------
extern "C" void kernel_launch(void* const* d_in, const int* in_sizes, int n_in,
                              void* d_out, int out_size, void* d_ws, size_t ws_size,
                              hipStream_t stream) {
    const float* x    = (const float*)d_in[0];
    const float* A    = (const float*)d_in[1];
    const float* Bm   = (const float*)d_in[2];
    const float* W    = (const float*)d_in[3];
    const float* bias = (const float*)d_in[4];
    float* out = (float*)d_out;

    unsigned short* zsp = (unsigned short*)d_ws;               // MDIM*RDIM bf16 = 1 MB
    unsigned short* Bbf = zsp + (size_t)MDIM * RDIM;           // NDIM*RDIM bf16 = 0.5 MB

    zk<<<MDIM / 128, 256, 0, stream>>>(x, A, zsp);
    bconv<<<(NDIM * RDIM) / (256 * 4), 256, 0, stream>>>(Bm, Bbf);
    gemmk<<<dim3(NDIM / BN, MDIM / BM), 256, 0, stream>>>(x, W, bias, zsp, Bbf, out);
}
